// BlockSparseAttentionOptimized_20143396618827
// MI455X (gfx1250) — matrix operand order repaired and checked by reading, not compiled, not run on hardware
//
#include <hip/hip_runtime.h>

typedef __attribute__((ext_vector_type(16))) _Float16 v16h;
typedef __attribute__((ext_vector_type(8)))  _Float16 v8h;
typedef __attribute__((ext_vector_type(8)))  float    v8f;

#define N_ATOM  4096
#define NHEADS  8
#define DHEAD   64
#define BLK     128
#define NROW    (N_ATOM / BLK)   // 32
#define BPS     4                // LOCAL_WINDOW / BLOCK_SIZE
#define KSTR    72               // padded LDS stride (halves) for K  [key][d]
#define VSTR    136              // padded LDS stride (halves) for Vt [d][key]
#define PSTR    136              // padded LDS stride (halves) for P  [qrow][key]

static __device__ __forceinline__ v16h cat8(v8h a, v8h b) {
  return __builtin_shufflevector(a, b, 0,1,2,3,4,5,6,7,8,9,10,11,12,13,14,15);
}

static __device__ __forceinline__ float exp2_hw(float x) {
  return __builtin_amdgcn_exp2f(x);   // v_exp_f32 (HW computes 2^x natively)
}

__global__ __launch_bounds__(256)
void block_sparse_attn_kernel(const float* __restrict__ q,
                              const float* __restrict__ k,
                              const float* __restrict__ v,
                              float* __restrict__ out)
{
  __shared__ __align__(16) _Float16 Ks[BLK * KSTR];     // K block, f16, [key][d]
  __shared__ __align__(16) _Float16 Vt[DHEAD * VSTR];   // V block transposed, [d][key]
  __shared__ __align__(16) _Float16 Pl[BLK * PSTR];     // P staging, [qrow(128)][key(128)]

  const int r    = blockIdx.x;        // query row-block
  const int h    = blockIdx.y;        // head
  const int tid  = threadIdx.x;
  const int lane = tid & 31;
  const int wv   = tid >> 5;          // wave id 0..7 -> owns 16 query rows
  const int mrow = lane & 15;
  const int hi   = lane >> 4;

  // softmax scale (1/sqrt(64)) with log2(e) folded in: S is produced in the
  // log2 domain so p = exp2(S - m) maps directly onto v_exp_f32.
  const float scale = 0.125f * 1.44269504088896340736f;

  // ---- Load Q tile (16x64) into WMMA A-layout registers, scale folded in ----
  // A operand (16x32) in this code's contiguous k order (B uses the same): lane l: M=l%16, hi=l/16, half t: K = hi*16 + t
  v16h aq[2];
  {
    const int qrow = r * BLK + wv * 16 + mrow;
    const float* qp = q + ((size_t)qrow * NHEADS + h) * DHEAD;
    #pragma unroll
    for (int kk = 0; kk < 2; ++kk) {
      v16h a;
      #pragma unroll
      for (int grp = 0; grp < 2; ++grp) {
        const float* src = qp + kk * 32 + hi * 16 + grp * 8;   // 32B aligned
        float4 f0 = *(const float4*)(src);
        float4 f1 = *(const float4*)(src + 4);
        a[grp*8 + 0] = (_Float16)(f0.x * scale);
        a[grp*8 + 1] = (_Float16)(f0.y * scale);
        a[grp*8 + 2] = (_Float16)(f0.z * scale);
        a[grp*8 + 3] = (_Float16)(f0.w * scale);
        a[grp*8 + 4] = (_Float16)(f1.x * scale);
        a[grp*8 + 5] = (_Float16)(f1.y * scale);
        a[grp*8 + 6] = (_Float16)(f1.z * scale);
        a[grp*8 + 7] = (_Float16)(f1.w * scale);
      }
      aq[kk] = a;
    }
  }

  // ---- Flash-attention state (per lane: 8 rows, row M = vi + 8*hi) ----
  const v8f zro = {0.f,0.f,0.f,0.f,0.f,0.f,0.f,0.f};
  v8f  o[4];                       // O accumulator, 16x64 as 4 C-tiles of 16x16
  o[0] = zro; o[1] = zro; o[2] = zro; o[3] = zro;
  float m8[8], s8[8];
  #pragma unroll
  for (int vi = 0; vi < 8; ++vi) { m8[vi] = -1e30f; s8[vi] = 0.f; }

  const int c0 = (r - BPS < 0) ? 0 : r - BPS;
  const int c1 = (r + BPS >= NROW) ? NROW - 1 : r + BPS;

  for (int c = c0; c <= c1; ++c) {
    // ---- Cooperative staging: K block -> Ks (f16), V block -> Vt (f16, transposed)
    #pragma unroll
    for (int j = 0; j < 8; ++j) {
      const int idx = tid + j * 256;          // 0..2047 : 128 keys x 16 float4-groups
      const int key = idx >> 4;
      const int d0  = (idx & 15) * 4;
      const size_t gbase = ((size_t)(c * BLK + key) * NHEADS + h) * DHEAD + d0;

      float4 kv = *(const float4*)(k + gbase);
      _Float16* kdp = &Ks[key * KSTR + d0];   // d0 multiple of 4 halves -> 8B aligned
      kdp[0] = (_Float16)kv.x; kdp[1] = (_Float16)kv.y;
      kdp[2] = (_Float16)kv.z; kdp[3] = (_Float16)kv.w;

      float4 vv = *(const float4*)(v + gbase);
      Vt[(d0 + 0) * VSTR + key] = (_Float16)vv.x;
      Vt[(d0 + 1) * VSTR + key] = (_Float16)vv.y;
      Vt[(d0 + 2) * VSTR + key] = (_Float16)vv.z;
      Vt[(d0 + 3) * VSTR + key] = (_Float16)vv.w;
    }
    __syncthreads();

    // ---- Prefetch next band block into L2/WGP$ while this block computes ----
    // 128 key rows x 256B (for this head) per tensor; 256 threads -> 128B each.
    if (c + 1 <= c1) {
      const int pk   = tid >> 1;
      const int poff = (tid & 1) * 32;
      const size_t pbase = ((size_t)((c + 1) * BLK + pk) * NHEADS + h) * DHEAD + poff;
      __builtin_prefetch(k + pbase, 0, 3);   // global_prefetch_b8
      __builtin_prefetch(v + pbase, 0, 3);
    }

    // ---- S = (Q*scale*log2e) . K^T : 8 subtiles of 16 keys, K-dim 64 = 2 x 32 ----
    v8f st[8];
    #pragma unroll
    for (int n = 0; n < 8; ++n) {
      v8f acc = zro;
      #pragma unroll
      for (int kk = 0; kk < 2; ++kk) {
        // B-layout (K=32 chans x N=16 keys): lane: N=key=n*16+mrow, half t: chan=kk*32+hi*16+t
        const _Float16* bp = &Ks[(n * 16 + mrow) * KSTR + kk * 32 + hi * 16];
        v8h b0 = *(const v8h*)(bp);
        v8h b1 = *(const v8h*)(bp + 8);
        acc = __builtin_amdgcn_wmma_f32_16x16x32_f16(
                false, aq[kk], false, cat8(b0, b1), (short)0, acc, false, false);
      }
      st[n] = acc;
    }

    // ---- Online softmax (log2 domain): row max across the 128 keys ----
    float alpha[8];
    #pragma unroll
    for (int vi = 0; vi < 8; ++vi) {
      float mm = st[0][vi];
      #pragma unroll
      for (int n = 1; n < 8; ++n) mm = fmaxf(mm, st[n][vi]);
      #pragma unroll
      for (int msk = 1; msk < 16; msk <<= 1)
        mm = fmaxf(mm, __shfl_xor(mm, msk, 32));
      const float mn = fmaxf(m8[vi], mm);
      alpha[vi] = exp2_hw(m8[vi] - mn);      // v_exp_f32, no extra mul
      m8[vi] = mn;
    }

    float rs[8];
    #pragma unroll
    for (int vi = 0; vi < 8; ++vi) rs[vi] = 0.f;
    #pragma unroll
    for (int n = 0; n < 8; ++n) {
      #pragma unroll
      for (int vi = 0; vi < 8; ++vi) {
        const float p = exp2_hw(st[n][vi] - m8[vi]);
        st[n][vi] = p;
        rs[vi] += p;
      }
    }
    #pragma unroll
    for (int vi = 0; vi < 8; ++vi) {
      #pragma unroll
      for (int msk = 1; msk < 16; msk <<= 1)
        rs[vi] += __shfl_xor(rs[vi], msk, 32);
      s8[vi] = s8[vi] * alpha[vi] + rs[vi];
    }
    #pragma unroll
    for (int dt = 0; dt < 4; ++dt)
      #pragma unroll
      for (int vi = 0; vi < 8; ++vi)
        o[dt][vi] *= alpha[vi];

    // ---- Stage P (C-layout f32) -> LDS f16 strip (wave-private rows) ----
    #pragma unroll
    for (int n = 0; n < 8; ++n)
      #pragma unroll
      for (int vi = 0; vi < 8; ++vi)
        Pl[(wv * 16 + vi + 8 * hi) * PSTR + n * 16 + mrow] = (_Float16)st[n][vi];

    // ---- O += P . V : 4 key-chunks of 32, 4 d-tiles of 16 ----
    #pragma unroll
    for (int kc = 0; kc < 4; ++kc) {
      // A-layout reload of P: halves t0..15 -> keys kc*32+hi*16+t
      const _Float16* pp = &Pl[(wv * 16 + mrow) * PSTR + kc * 32 + hi * 16];
      v8h p0 = *(const v8h*)(pp);
      v8h p1 = *(const v8h*)(pp + 8);
      const v16h pa = cat8(p0, p1);
      #pragma unroll
      for (int dt = 0; dt < 4; ++dt) {
        // B-layout: lane: N = dcol = dt*16+mrow, half t: key = kc*32 + hi*16 + t
        const _Float16* vp = &Vt[(dt * 16 + mrow) * VSTR + kc * 32 + hi * 16];
        v8h b0 = *(const v8h*)(vp);
        v8h b1 = *(const v8h*)(vp + 8);
        o[dt] = __builtin_amdgcn_wmma_f32_16x16x32_f16(
                  false, pa, false, cat8(b0, b1), (short)0, o[dt], false, false);
      }
    }
    __syncthreads();   // protect Ks/Vt before next block's staging
  }

  // ---- Normalize and write O ----
  float inv[8];
  #pragma unroll
  for (int vi = 0; vi < 8; ++vi) inv[vi] = 1.0f / s8[vi];
  #pragma unroll
  for (int dt = 0; dt < 4; ++dt) {
    #pragma unroll
    for (int vi = 0; vi < 8; ++vi) {
      const int row = r * BLK + wv * 16 + vi + 8 * hi;
      out[((size_t)row * NHEADS + h) * DHEAD + dt * 16 + mrow] = o[dt][vi] * inv[vi];
    }
  }
}

extern "C" void kernel_launch(void* const* d_in, const int* in_sizes, int n_in,
                              void* d_out, int out_size, void* d_ws, size_t ws_size,
                              hipStream_t stream) {
  const float* q = (const float*)d_in[0];
  const float* k = (const float*)d_in[1];
  const float* v = (const float*)d_in[2];
  // d_in[3] = pair_bias: unused by the reference forward (never read).
  float* out = (float*)d_out;
  (void)in_sizes; (void)n_in; (void)out_size; (void)d_ws; (void)ws_size;

  dim3 grid(NROW, NHEADS);
  dim3 block(256);
  block_sparse_attn_kernel<<<grid, block, 0, stream>>>(q, k, v, out);
}
